// CatAndNonLinearMultiary_87557203296994
// MI455X (gfx1250) — matrix operand order repaired and checked by reading, not compiled, not run on hardware
//
#include <hip/hip_runtime.h>
#include <hip/hip_bf16.h>

typedef __attribute__((ext_vector_type(16))) _Float16 v16h;
typedef __attribute__((ext_vector_type(8)))  _Float16 v8h;
typedef __attribute__((ext_vector_type(4)))  _Float16 v4h;
typedef __attribute__((ext_vector_type(8)))  float    v8f;

#define DM    128     // embedding dim
#define TWOD  256
#define BUFS  136     // halves per tree-buffer row (272B: 68 banks -> skewed, conflict-free col reads)
#define HS    264     // halves per H row (528B: 132 banks -> skewed)
#define YS    132     // floats per Y row (same 528B region as H)

static __device__ __forceinline__ v16h cat8(v8h lo, v8h hi) {
  return __builtin_shufflevector(lo, hi, 0,1,2,3,4,5,6,7,8,9,10,11,12,13,14,15);
}

// ---------------------------------------------------------------------------
// Prep: transpose + convert weights to f16, N-major, into global scratch.
// wt1[n][k] = W1[k][n]  (n,k in [0,256));  wt2[n][k] = W2[k][n] (n<128,k<256)
// Every WMMA B-fragment becomes two 16B runs per lane (L2-hot).
// ---------------------------------------------------------------------------
__global__ __launch_bounds__(256) void prep_weights_kernel(
    const float* __restrict__ W1, const float* __restrict__ W2,
    _Float16* __restrict__ wt1, _Float16* __restrict__ wt2) {
  int id = blockIdx.x * 256 + threadIdx.x;
  if (id < 256 * 256) {
    int n = id >> 8, k = id & 255;
    wt1[n * 256 + k] = (_Float16)W1[k * 256 + n];
  } else if (id < 256 * 256 + 128 * 256) {
    int j = id - 256 * 256;
    int n = j >> 8, k = j & 255;
    wt2[n * 256 + k] = (_Float16)W2[k * 128 + n];
  }
}

// ---------------------------------------------------------------------------
// Main: one block per ragged item, 8 wave32s. Level-synchronized binary tree
// reduction; pair-MLP as WMMA f16->f32 GEMMs. Each wave owns GEMM1 column
// tiles {wave, wave+8} with its W1 fragments resident in VGPRs for the whole
// kernel, and GEMM2 column tile {wave}.
// ---------------------------------------------------------------------------
__global__ __launch_bounds__(256) void tree_mlp_reduce_kernel(
    const float* __restrict__ args,
    const int*   __restrict__ limits,
    const float* __restrict__ b1,
    const float* __restrict__ b2,
    const float* __restrict__ ln_w,
    const float* __restrict__ ln_b,
    const _Float16* __restrict__ wt1,
    const _Float16* __restrict__ wt2,
    float* __restrict__ out)
{
  __shared__ _Float16 sBuf[256 * BUFS];                  // 69632 B tree buffer (f16)
  __shared__ __align__(16) char sHY[64 * HS * 2];        // 33792 B: H (f16) / Y (f32) overlay
  __shared__ float sB1[TWOD], sB2[DM], sLnW[DM], sLnB[DM];

  _Float16* Hh = (_Float16*)sHY;
  float*    Yf = (float*)sHY;

  const int tid    = threadIdx.x;
  const int wave   = tid >> 5;
  const int lane   = tid & 31;
  const int laneLo = lane & 15;
  const int hiA    = (lane >= 16) ? 8  : 0;   // A-frag K offset / C-row offset
  const int hiB    = (lane >= 16) ? 16 : 0;   // unused; B-frag K offset is hiA

  const int start = limits[blockIdx.x];
  const int L     = limits[blockIdx.x + 1] - start;

  if (L == 1) {  // tree root is the input element itself (keep f32 precision)
    if (tid < DM) out[blockIdx.x * DM + tid] = args[(size_t)start * DM + tid];
    return;
  }

  // ---- kernel-lifetime W1 fragments: nt0 = wave, nt1 = wave + 8 ----
  v16h w1f[16];
  {
    const _Float16* wc0 = wt1 + (wave * 16 + laneLo) * 256;
    const _Float16* wc1 = wt1 + ((wave + 8) * 16 + laneLo) * 256;
    #pragma unroll
    for (int kt = 0; kt < 8; ++kt) {
      w1f[kt]     = cat8(*(const v8h*)(wc0 + kt * 32 + hiA),
                         *(const v8h*)(wc0 + kt * 32 + hiA + 16));
      w1f[kt + 8] = cat8(*(const v8h*)(wc1 + kt * 32 + hiA),
                         *(const v8h*)(wc1 + kt * 32 + hiA + 16));
    }
  }

  // biases / LN params to LDS
  for (int i = tid; i < TWOD; i += 256) sB1[i] = b1[i];
  if (tid < DM) { sB2[tid] = b2[tid]; sLnW[tid] = ln_w[tid]; sLnB[tid] = ln_b[tid]; }

  // sequence rows -> f16 tree buffer
  for (int c = tid; c < L * 32; c += 256) {
    int row = c >> 5, q = (c & 31) << 2;
    float4 v = *(const float4*)(args + (size_t)(start + row) * DM + q);
    v4h h = { (_Float16)v.x, (_Float16)v.y, (_Float16)v.z, (_Float16)v.w };
    *(v4h*)(sBuf + row * BUFS + q) = h;
  }
  __syncthreads();

  const float bias0 = sB1[wave * 16 + laneLo];          // GEMM1 col tile nt0
  const float bias1 = sB1[wave * 16 + laneLo + 128];    // GEMM1 col tile nt1
  const float bias2 = sB2[wave * 16 + laneLo];          // GEMM2 col tile
  const _Float16* wc2 = wt2 + (wave * 16 + laneLo) * 256;

  int l = L;
  while (l > 1) {
    const int pairs = l >> 1;
    const int odd   = l & 1;
    const bool finalLevel = (l == 2);

    for (int p0 = 0; p0 < pairs; p0 += 64) {
      int chunkP = pairs - p0; if (chunkP > 64) chunkP = 64;
      const int mtiles = (chunkP + 15) >> 4;

      // ---- GEMM1: H[chunk,256] = relu(X @ W1 + b1); 2 col-tiles per wave ----
      for (int mt = 0; mt < mtiles; ++mt) {
        int p = p0 + mt * 16 + laneLo;
        if (p > pairs - 1) p = pairs - 1;            // duplicate tail pair (rows discarded)
        v8f acc0 = { bias0, bias0, bias0, bias0, bias0, bias0, bias0, bias0 };
        v8f acc1 = { bias1, bias1, bias1, bias1, bias1, bias1, bias1, bias1 };
        #pragma unroll
        for (int kt = 0; kt < 8; ++kt) {
          const int k0 = kt * 32 + hiA;              // A elems e=0..7 -> K=k0..k0+7
          const int k1 = k0 + 16;                    // A elems e=8..15 -> K=k1..k1+7
          v8h alo = *(const v8h*)(sBuf + (2 * p + (k0 >> 7)) * BUFS + (k0 & 127));
          v8h ahi = *(const v8h*)(sBuf + (2 * p + (k1 >> 7)) * BUFS + (k1 & 127));
          v16h a = cat8(alo, ahi);
          acc0 = __builtin_amdgcn_wmma_f32_16x16x32_f16(false, a, false, w1f[kt],
                                                        (short)0, acc0, false, false);
          acc1 = __builtin_amdgcn_wmma_f32_16x16x32_f16(false, a, false, w1f[kt + 8],
                                                        (short)0, acc1, false, false);
        }
        #pragma unroll
        for (int r = 0; r < 8; ++r) {
          float v0 = acc0[r] > 0.0f ? acc0[r] : 0.0f;   // ReLU
          float v1 = acc1[r] > 0.0f ? acc1[r] : 0.0f;
          _Float16* hr = Hh + (mt * 16 + hiA + r) * HS + laneLo;
          hr[wave * 16]       = (_Float16)v0;
          hr[wave * 16 + 128] = (_Float16)v1;
        }
      }
      __syncthreads();

      // ---- GEMM2: Y = H @ W2 + b2; nt = wave, 2 row-tiles per B-fragment ----
      const int mtP = (mtiles + 1) & ~1;             // even (garbage rows never read)
      v8f acc2[4];
      for (int mt0 = 0; mt0 < mtP; mt0 += 2) {
        v8f accA = { bias2, bias2, bias2, bias2, bias2, bias2, bias2, bias2 };
        v8f accB = accA;
        const int r0 = mt0 * 16 + laneLo;
        const int r1 = r0 + 16;
        #pragma unroll
        for (int kt = 0; kt < 8; ++kt) {
          v16h bm = cat8(*(const v8h*)(wc2 + kt * 32 + hiA),
                         *(const v8h*)(wc2 + kt * 32 + hiA + 16));
          const int k0 = kt * 32 + hiA;
          v16h aA = cat8(*(const v8h*)(Hh + r0 * HS + k0),
                         *(const v8h*)(Hh + r0 * HS + k0 + 16));
          v16h aB = cat8(*(const v8h*)(Hh + r1 * HS + k0),
                         *(const v8h*)(Hh + r1 * HS + k0 + 16));
          accA = __builtin_amdgcn_wmma_f32_16x16x32_f16(false, aA, false, bm,
                                                        (short)0, accA, false, false);
          accB = __builtin_amdgcn_wmma_f32_16x16x32_f16(false, aB, false, bm,
                                                        (short)0, accB, false, false);
        }
        acc2[mt0] = accA; acc2[mt0 + 1] = accB;
      }
      __syncthreads();                               // all H reads done before Y overlays it
      for (int mt = 0; mt < mtP; ++mt) {
        #pragma unroll
        for (int r = 0; r < 8; ++r)
          Yf[(mt * 16 + hiA + r) * YS + wave * 16 + laneLo] = acc2[mt][r];
      }
      __syncthreads();

      // ---- LayerNorm per row (one wave32 per row) + in-place writeback ----
      for (int row = wave; row < chunkP; row += 8) {
        const float* yr = Yf + row * YS;
        float y[4];
        #pragma unroll
        for (int q = 0; q < 4; ++q) y[q] = yr[lane + 32 * q];
        float s  = y[0] + y[1] + y[2] + y[3];
        float ss = y[0]*y[0] + y[1]*y[1] + y[2]*y[2] + y[3]*y[3];
        #pragma unroll
        for (int off = 16; off > 0; off >>= 1) {
          s  += __shfl_xor(s,  off, 32);
          ss += __shfl_xor(ss, off, 32);
        }
        const float mean = s * (1.0f / 128.0f);
        const float var  = ss * (1.0f / 128.0f) - mean * mean;
        const float inv  = rsqrtf(var + 1e-5f);
        const int j = p0 + row;                      // global pair index this level
        #pragma unroll
        for (int q = 0; q < 4; ++q) {
          const int col = lane + 32 * q;
          const float o = (y[q] - mean) * inv * sLnW[col] + sLnB[col];
          if (finalLevel) out[blockIdx.x * DM + col] = o;     // root -> f32 output
          else sBuf[(odd + j) * BUFS + col] = (_Float16)o;    // next-level row
        }
      }
      __syncthreads();
    } // chunk loop

    if (odd) {                                       // leftover becomes row 0 of next level
      if (tid < DM) sBuf[tid] = sBuf[(l - 1) * BUFS + tid];
    }
    __syncthreads();
    l = pairs + odd;
  }
}

// ---------------------------------------------------------------------------
extern "C" void kernel_launch(void* const* d_in, const int* in_sizes, int n_in,
                              void* d_out, int out_size, void* d_ws, size_t ws_size,
                              hipStream_t stream) {
  const float* args   = (const float*)d_in[0];
  const int*   limits = (const int*)d_in[1];
  const float* W1     = (const float*)d_in[2];
  const float* b1     = (const float*)d_in[3];
  const float* W2     = (const float*)d_in[4];
  const float* b2     = (const float*)d_in[5];
  const float* lnw    = (const float*)d_in[6];
  const float* lnb    = (const float*)d_in[7];

  _Float16* wt1 = (_Float16*)d_ws;                 // 256*256 f16
  _Float16* wt2 = wt1 + 256 * 256;                 // 128*256 f16  (total 192 KB, L2-hot)

  const int B = in_sizes[1] - 1;                   // limits has B+1 entries
  const int prepElems = 256 * 256 + 128 * 256;

  prep_weights_kernel<<<(prepElems + 255) / 256, 256, 0, stream>>>(W1, W2, wt1, wt2);
  tree_mlp_reduce_kernel<<<B, 256, 0, stream>>>(args, limits, b1, b2, lnw, lnb,
                                                wt1, wt2, (float*)d_out);
}
